// MultiHeadAttention_9990093931173
// MI455X (gfx1250) — compile-verified, matrix operand order checked by reading, not run on hardware
//
#include <hip/hip_runtime.h>

#define EMB 768
#define NH 12
#define HD 64
#define BATCH 8
#define SEQ 1024
#define MTOT (BATCH * SEQ)   // 8192 rows

typedef __attribute__((ext_vector_type(8)))  __bf16 v8bf;
typedef __attribute__((ext_vector_type(16))) __bf16 v16bf;
typedef __attribute__((ext_vector_type(8)))  float  v8f;
typedef __attribute__((ext_vector_type(4)))  unsigned int u32x4;
typedef __attribute__((ext_vector_type(4)))  int i32x4;
typedef __attribute__((ext_vector_type(8)))  int i32x8;
typedef unsigned short ushort_t;

// ---- helpers -------------------------------------------------------------

__device__ __forceinline__ ushort_t f2bf(float f) {
    union { float f; unsigned int u; } c;
    c.f = f;
    unsigned int u = c.u;
    u += 0x7FFFu + ((u >> 16) & 1u);   // round to nearest even
    return (ushort_t)(u >> 16);
}

// Lane's 16 A/B-fragment elements = two contiguous 8-elem (16B) chunks at p, p+16.
__device__ __forceinline__ v16bf frag_ld(const ushort_t* p) {
    v8bf lo = *(const v8bf*)(p);
    v8bf hi = *(const v8bf*)(p + 16);
    return __builtin_shufflevector(lo, hi, 0, 1, 2, 3, 4, 5, 6, 7,
                                   8, 9, 10, 11, 12, 13, 14, 15);
}

__device__ __forceinline__ v8f wmma_bf16(v16bf a, v16bf b, v8f c) {
    return __builtin_amdgcn_wmma_f32_16x16x32_bf16(false, a, false, b,
                                                   (short)0, c, false, false);
}

// ---- Tensor Data Mover: 2D tile (16-bit elements) global -> LDS ----------
// D# per CDNA5 ISA. Groups 2/3 zero (2D tensor).
__device__ __forceinline__ void tdm_load_2d(void* lds_dst, const void* gsrc,
                                            unsigned int tile_w,      // elems/row
                                            unsigned int tile_h,      // rows
                                            unsigned long long stride_el,
                                            unsigned long long dim0_el,
                                            unsigned int dim1_rows) {
    unsigned long long ga = (unsigned long long)(size_t)gsrc;
    unsigned int lds_off = (unsigned int)(size_t)lds_dst;   // low 32b = LDS byte addr
    u32x4 g0;
    g0[0] = 1u;                                             // count=1, user mode
    g0[1] = lds_off;                                        // lds_addr
    g0[2] = (unsigned int)ga;                               // global_addr[31:0]
    g0[3] = (unsigned int)((ga >> 32) & 0x01FFFFFFu) | (2u << 30);  // [56:32] | type=2
    i32x8 g1;
    g1[0] = 0x00010000;                                     // data_size=1 (2B), no mcast
    g1[1] = (int)((dim0_el & 0xFFFFu) << 16);               // tensor_dim0[15:0] @bit48
    g1[2] = (int)(((dim0_el >> 16) & 0xFFFFu) |             // tensor_dim0[31:16]
                  ((dim1_rows & 0xFFFFu) << 16));           // tensor_dim1[15:0]
    g1[3] = (int)(((dim1_rows >> 16) & 0xFFFFu) |           // tensor_dim1[31:16]
                  ((tile_w & 0xFFFFu) << 16));              // tile_dim0
    g1[4] = (int)(tile_h & 0xFFFFu);                        // tile_dim1 (tile_dim2=0)
    g1[5] = (int)(unsigned int)(stride_el & 0xFFFFFFFFu);   // dim0_stride[31:0]
    g1[6] = (int)((stride_el >> 32) & 0xFFFFu);             // dim0_stride[47:32]
    g1[7] = 0;
    i32x4 z4 = {0, 0, 0, 0};
#if __has_include(<hip/amd_detail/amd_gfx1250_TDM.h>)
    i32x8 z8 = {0, 0, 0, 0, 0, 0, 0, 0};
    __builtin_amdgcn_tensor_load_to_lds(g0, g1, z4, z4, z8, 0);
#else
    __builtin_amdgcn_tensor_load_to_lds(g0, g1, z4, z4, 0);
#endif
}

// ---- conversion kernels --------------------------------------------------

__global__ void cvt_f32_bf16(const float* __restrict__ in,
                             ushort_t* __restrict__ out, int n) {
    int i = blockIdx.x * blockDim.x + threadIdx.x;
    if (i < n) out[i] = f2bf(in[i]);
}

// in [rows][cols] f32  ->  out [cols][rows] bf16   (weights: [K][N] -> [N][K])
__global__ void cvt_transpose(const float* __restrict__ in,
                              ushort_t* __restrict__ out, int rows, int cols) {
    int i = blockIdx.x * blockDim.x + threadIdx.x;
    if (i < rows * cols) {
        int r = i / cols, c = i % cols;
        out[(size_t)c * rows + r] = f2bf(in[i]);
    }
}

// ---- GEMM 1: qkv = x @ w_qkv, scatter to Q(scaled)/K/V^T -----------------
// A: [8192][768] bf16 row-major.  Bt: [2304][768] bf16 (N-major).
// Block tile 128x128, 8 waves 2(M) x 4(N), wave tile 64x32, K-tile 64 via TDM.

__global__ void __launch_bounds__(256) gemm_qkv(
    const ushort_t* __restrict__ A, const ushort_t* __restrict__ Bt,
    ushort_t* __restrict__ Qb, ushort_t* __restrict__ Kb,
    ushort_t* __restrict__ Vt) {
    __shared__ __align__(16) ushort_t sA[128 * 64];
    __shared__ __align__(16) ushort_t sB[128 * 64];
    const int tid = threadIdx.x, lane = tid & 31, w = tid >> 5;
    const int wm = w & 1, wn = w >> 1;
    const int mBase = blockIdx.y * 128, nBase = blockIdx.x * 128;
    const int ln = lane & 15, half = lane >> 4;

    v8f zacc = {};
    v8f acc[4][2];
#pragma unroll
    for (int i = 0; i < 4; ++i)
#pragma unroll
        for (int j = 0; j < 2; ++j) acc[i][j] = zacc;

    for (int k0 = 0; k0 < EMB; k0 += 64) {
        if (w == 0) {
            tdm_load_2d(sA, A + (size_t)mBase * EMB + k0, 64, 128, EMB, EMB, MTOT);
            tdm_load_2d(sB, Bt + (size_t)nBase * EMB + k0, 64, 128, EMB, EMB, 3 * EMB);
            __builtin_amdgcn_s_wait_tensorcnt(0);
        }
        __syncthreads();
#pragma unroll
        for (int ks = 0; ks < 64; ks += 32) {
            const int cb = ks + half * 8;
            v16bf af[4], bfg[2];
#pragma unroll
            for (int i = 0; i < 4; ++i)
                af[i] = frag_ld(sA + (wm * 64 + i * 16 + ln) * 64 + cb);
#pragma unroll
            for (int j = 0; j < 2; ++j)
                bfg[j] = frag_ld(sB + (wn * 32 + j * 16 + ln) * 64 + cb);
#pragma unroll
            for (int i = 0; i < 4; ++i)
#pragma unroll
                for (int j = 0; j < 2; ++j)
                    acc[i][j] = wmma_bf16(af[i], bfg[j], acc[i][j]);
        }
        __syncthreads();
    }

    // scatter epilogue: n -> {q,k,v}, head, d
#pragma unroll
    for (int j = 0; j < 2; ++j) {
        int n = nBase + wn * 32 + j * 16 + ln;
        int sel = n / EMB;             // 0=q 1=k 2=v
        int nn = n - sel * EMB;
        int h = nn / HD, d = nn % HD;
#pragma unroll
        for (int i = 0; i < 4; ++i) {
#pragma unroll
            for (int r = 0; r < 8; ++r) {
                int row = mBase + wm * 64 + i * 16 + r + 8 * half;
                int b = row >> 10, s = row & 1023;
                int bh = b * NH + h;
                float v = acc[i][j][r];
                if (sel == 0)
                    Qb[((size_t)bh * SEQ + s) * HD + d] = f2bf(v * 0.125f); // fold 1/sqrt(64)
                else if (sel == 1)
                    Kb[((size_t)bh * SEQ + s) * HD + d] = f2bf(v);
                else
                    Vt[((size_t)bh * HD + d) * SEQ + s] = f2bf(v);
            }
        }
    }
}

// ---- Flash attention: per (head, 128-query block), waves own 16 queries --

__global__ void __launch_bounds__(256) attn_kernel(
    const ushort_t* __restrict__ Qb, const ushort_t* __restrict__ Kb,
    const ushort_t* __restrict__ Vt, ushort_t* __restrict__ attn_out) {
    __shared__ __align__(16) ushort_t kT[64 * 64];       // [key][d]
    __shared__ __align__(16) ushort_t vS[64 * 64];       // [d][key]
    __shared__ __align__(16) ushort_t pT[8][16 * 64];    // per-wave P: [q][key]
    const int tid = threadIdx.x, lane = tid & 31, w = tid >> 5;
    const int bh = blockIdx.x >> 3;            // 0..95
    const int qBase = (blockIdx.x & 7) * 128;
    const int ln = lane & 15, half = lane >> 4;

    // Q fragments for this wave's 16 queries (D=64 -> 2 k-steps), pre-scaled
    v16bf qf[2];
    {
        const ushort_t* qrow = Qb + ((size_t)bh * SEQ + qBase + w * 16 + ln) * HD;
#pragma unroll
        for (int kk = 0; kk < 2; ++kk)
            qf[kk] = frag_ld(qrow + kk * 32 + half * 8);
    }

    float mrun[8], lrun[8];
#pragma unroll
    for (int r = 0; r < 8; ++r) { mrun[r] = -1e30f; lrun[r] = 0.f; }
    v8f zacc = {};
    v8f oacc[4];
#pragma unroll
    for (int dt = 0; dt < 4; ++dt) oacc[dt] = zacc;

    for (int kb = 0; kb < SEQ; kb += 64) {
        // TDM stage: K block [64 keys][64 d], V^T block [64 d][64 keys]
        if (w == 0) {
            tdm_load_2d(kT, Kb + ((size_t)bh * SEQ + kb) * HD, 64, 64, HD, HD, SEQ);
            tdm_load_2d(vS, Vt + (size_t)bh * HD * SEQ + kb, 64, 64, SEQ, SEQ, HD);
            __builtin_amdgcn_s_wait_tensorcnt(0);
        }
        __syncthreads();

        // scores: 16 queries x 64 keys
        v8f sc[4];
#pragma unroll
        for (int nt = 0; nt < 4; ++nt) {
            sc[nt] = zacc;
#pragma unroll
            for (int kk = 0; kk < 2; ++kk) {
                v16bf bk = frag_ld(kT + (nt * 16 + ln) * 64 + kk * 32 + half * 8);
                sc[nt] = wmma_bf16(qf[kk], bk, sc[nt]);
            }
        }

        // online softmax: rows r+8*half live across the 16 lanes of this half
#pragma unroll
        for (int r = 0; r < 8; ++r) {
            float m = fmaxf(fmaxf(sc[0][r], sc[1][r]), fmaxf(sc[2][r], sc[3][r]));
            m = fmaxf(m, __shfl_xor(m, 1, 32));
            m = fmaxf(m, __shfl_xor(m, 2, 32));
            m = fmaxf(m, __shfl_xor(m, 4, 32));
            m = fmaxf(m, __shfl_xor(m, 8, 32));
            float nm = fmaxf(mrun[r], m);
            float alpha = __expf(mrun[r] - nm);
            mrun[r] = nm;
            float rs = 0.f;
#pragma unroll
            for (int nt = 0; nt < 4; ++nt) {
                float p = __expf(sc[nt][r] - nm);
                sc[nt][r] = p;
                rs += p;
            }
            rs += __shfl_xor(rs, 1, 32);
            rs += __shfl_xor(rs, 2, 32);
            rs += __shfl_xor(rs, 4, 32);
            rs += __shfl_xor(rs, 8, 32);
            lrun[r] = lrun[r] * alpha + rs;
#pragma unroll
            for (int dt = 0; dt < 4; ++dt) oacc[dt][r] *= alpha;
        }

        // C-fragment -> A-fragment layout change via wave-private LDS
#pragma unroll
        for (int nt = 0; nt < 4; ++nt)
#pragma unroll
            for (int r = 0; r < 8; ++r)
                pT[w][(r + 8 * half) * 64 + nt * 16 + ln] = f2bf(sc[nt][r]);

        // O += P @ V   (2 k-steps x 4 d-tiles)
#pragma unroll
        for (int kk = 0; kk < 2; ++kk) {
            v16bf pf = frag_ld(pT[w] + ln * 64 + kk * 32 + half * 8);
#pragma unroll
            for (int dt = 0; dt < 4; ++dt) {
                v16bf vf = frag_ld(vS + (dt * 16 + ln) * 64 + kk * 32 + half * 8);
                oacc[dt] = wmma_bf16(pf, vf, oacc[dt]);
            }
        }
        __syncthreads();
    }

    // normalize and write to [B,S,E] bf16
    const int b = bh / NH, h = bh % NH;
#pragma unroll
    for (int r = 0; r < 8; ++r) {
        float inv = lrun[r] > 0.f ? 1.0f / lrun[r] : 0.f;
        int srow = qBase + w * 16 + r + 8 * half;
        size_t base = ((size_t)b * SEQ + srow) * EMB + h * HD;
#pragma unroll
        for (int dt = 0; dt < 4; ++dt)
            attn_out[base + dt * 16 + ln] = f2bf(oacc[dt][r] * inv);
    }
}

// ---- GEMM 2: out = attn @ w_out + b_out  (f32 output) --------------------

__global__ void __launch_bounds__(256) gemm_out(
    const ushort_t* __restrict__ A, const ushort_t* __restrict__ Bt,
    const float* __restrict__ bias, float* __restrict__ out) {
    __shared__ __align__(16) ushort_t sA[128 * 64];
    __shared__ __align__(16) ushort_t sB[128 * 64];
    const int tid = threadIdx.x, lane = tid & 31, w = tid >> 5;
    const int wm = w & 1, wn = w >> 1;
    const int mBase = blockIdx.y * 128, nBase = blockIdx.x * 128;
    const int ln = lane & 15, half = lane >> 4;

    v8f zacc = {};
    v8f acc[4][2];
#pragma unroll
    for (int i = 0; i < 4; ++i)
#pragma unroll
        for (int j = 0; j < 2; ++j) acc[i][j] = zacc;

    for (int k0 = 0; k0 < EMB; k0 += 64) {
        if (w == 0) {
            tdm_load_2d(sA, A + (size_t)mBase * EMB + k0, 64, 128, EMB, EMB, MTOT);
            tdm_load_2d(sB, Bt + (size_t)nBase * EMB + k0, 64, 128, EMB, EMB, EMB);
            __builtin_amdgcn_s_wait_tensorcnt(0);
        }
        __syncthreads();
#pragma unroll
        for (int ks = 0; ks < 64; ks += 32) {
            const int cb = ks + half * 8;
            v16bf af[4], bfg[2];
#pragma unroll
            for (int i = 0; i < 4; ++i)
                af[i] = frag_ld(sA + (wm * 64 + i * 16 + ln) * 64 + cb);
#pragma unroll
            for (int j = 0; j < 2; ++j)
                bfg[j] = frag_ld(sB + (wn * 32 + j * 16 + ln) * 64 + cb);
#pragma unroll
            for (int i = 0; i < 4; ++i)
#pragma unroll
                for (int j = 0; j < 2; ++j)
                    acc[i][j] = wmma_bf16(af[i], bfg[j], acc[i][j]);
        }
        __syncthreads();
    }

#pragma unroll
    for (int j = 0; j < 2; ++j) {
        int n = nBase + wn * 32 + j * 16 + ln;
        float bv = bias[n];
#pragma unroll
        for (int i = 0; i < 4; ++i) {
#pragma unroll
            for (int r = 0; r < 8; ++r) {
                int row = mBase + wm * 64 + i * 16 + r + 8 * half;
                out[(size_t)row * EMB + n] = acc[i][j][r] + bv;
            }
        }
    }
}

// ---- host launch ---------------------------------------------------------

extern "C" void kernel_launch(void* const* d_in, const int* in_sizes, int n_in,
                              void* d_out, int out_size, void* d_ws, size_t ws_size,
                              hipStream_t stream) {
    const float* x     = (const float*)d_in[0];
    const float* w_qkv = (const float*)d_in[1];
    const float* w_out = (const float*)d_in[2];
    const float* b_out = (const float*)d_in[3];
    float* out = (float*)d_out;

    char* ws = (char*)d_ws;
    size_t off = 0;
    auto carve = [&](size_t bytes) {
        void* p = ws + off;
        off += (bytes + 255) & ~(size_t)255;
        return p;
    };
    const size_t XBYTES = (size_t)MTOT * EMB * sizeof(ushort_t);   // 12.6 MB
    ushort_t* x_bf  = (ushort_t*)carve(XBYTES);                    // reused as attn output
    ushort_t* wqkvT = (ushort_t*)carve((size_t)3 * EMB * EMB * sizeof(ushort_t));
    ushort_t* woutT = (ushort_t*)carve((size_t)EMB * EMB * sizeof(ushort_t));
    ushort_t* Qb    = (ushort_t*)carve(XBYTES);
    ushort_t* Kb    = (ushort_t*)carve(XBYTES);
    ushort_t* Vtb   = (ushort_t*)carve(XBYTES);

    const int nx = MTOT * EMB;
    cvt_f32_bf16<<<(nx + 255) / 256, 256, 0, stream>>>(x, x_bf, nx);
    cvt_transpose<<<(EMB * 3 * EMB + 255) / 256, 256, 0, stream>>>(w_qkv, wqkvT, EMB, 3 * EMB);
    cvt_transpose<<<(EMB * EMB + 255) / 256, 256, 0, stream>>>(w_out, woutT, EMB, EMB);

    gemm_qkv<<<dim3(3 * EMB / 128, MTOT / 128), 256, 0, stream>>>(x_bf, wqkvT, Qb, Kb, Vtb);

    // attention output aliases x_bf (dead after gemm_qkv; stream-ordered)
    attn_kernel<<<dim3(BATCH * NH * (SEQ / 128)), 256, 0, stream>>>(Qb, Kb, Vtb, x_bf);

    gemm_out<<<dim3(EMB / 128, MTOT / 128), 256, 0, stream>>>(x_bf, woutT, b_out, out);
}
